// Piglet_78408922956333
// MI455X (gfx1250) — matrix operand order repaired and checked by reading, not compiled, not run on hardware
//
#include <hip/hip_runtime.h>
#include <hip/hip_bf16.h>
#include <math.h>

// ---------------------------------------------------------------------------
// Signed-graph GNN on MI455X (gfx1250, wave32).
//   - Edge aggregation: coalesced gather + global_atomic_add_f32 scatter
//     (memory/atomic bound; ~1.6 GB traffic, agg targets are L2-resident).
//   - Dense stages: fused wmean-normalize + concat + GEMM + bias + tanh,
//     one wave per 16-node tile using v_wmma_f32_16x16x32_f16.
//   - Weights pre-packed once per launch into per-lane f16 B-fragment layout
//     so each fragment is two coalesced global_load_b128 per lane.
// ---------------------------------------------------------------------------

typedef __attribute__((ext_vector_type(16))) _Float16 v16h;
typedef __attribute__((ext_vector_type(8)))  float    v8f;

__device__ __forceinline__ v8f wmma_f32_f16(v16h a, v16h b, v8f c) {
  // 8 args: (neg_a, A, neg_b, B, c_mod, C, reuse_a, reuse_b)
  return __builtin_amdgcn_wmma_f32_16x16x32_f16(
      false, a, false, b, (short)0, c, false, false);
}

// A fragment K index map (16x32 f16 A layout): half i -> K = k0 + i + (i>=8?8:0) + khi*8
__device__ __forceinline__ int a_kmap(int k0, int i, int khi) {
  return k0 + i + (i >= 8 ? 8 : 0) + khi * 8;
}

// Packed B fragment: P[((frag*32 + lane)*16 .. +15], 32B-aligned per lane.
__device__ __forceinline__ v16h load_packed_frag(const _Float16* __restrict__ P,
                                                 int frag, int lane) {
  return *(const v16h*)(P + ((size_t)frag * 32 + lane) * 16);
}

// ---------------------------------------------------------------------------
// Pack a (K x ncols) f32 weight matrix into WMMA B-fragment f16 layout.
// Fragment layout (32x16 KxN, f16): lane&15 = N column, lane>=16 selects the
// K half; half i holds K = kstep*32 + i + (i>=8?8:0) + khi*8.
// frag = kstep * (ncols/16) + ntile.  One thread per output half.
// ---------------------------------------------------------------------------
__global__ void pack_w_kernel(const float* __restrict__ W,
                              _Float16* __restrict__ P, int K, int ncols) {
  int t = blockIdx.x * blockDim.x + threadIdx.x;
  if (t >= K * ncols) return;
  int i      = t & 15;
  int lane   = (t >> 4) & 31;
  int frag   = t >> 9;
  int ntiles = ncols >> 4;
  int kstep  = frag / ntiles;
  int ntile  = frag - kstep * ntiles;
  int m      = lane & 15;
  int khi    = lane >> 4;
  int k      = kstep * 32 + i + (i >= 8 ? 8 : 0) + khi * 8;
  int ncol   = ntile * 16 + m;
  P[t] = (_Float16)W[k * ncols + ncol];
}

// ---------------------------------------------------------------------------
// Degree (weighted in-degree) accumulation: d[dst] += w
// ---------------------------------------------------------------------------
__global__ void degree_kernel(const int* __restrict__ ei,
                              const float* __restrict__ w,
                              float* __restrict__ d, int E) {
  int t = blockIdx.x * blockDim.x + threadIdx.x;
  if (t < E) atomicAdd(&d[ei[E + t]], w[t]);
}

// ---------------------------------------------------------------------------
// Weighted feature scatter: out[dst,c] += feat[src,c] * w[e], 64 channels/edge.
// 64 consecutive threads = one edge -> 256B coalesced gather + atomic burst.
// ---------------------------------------------------------------------------
__global__ void scatter_kernel(const int* __restrict__ ei,
                               const float* __restrict__ w,
                               const float* __restrict__ feat,
                               float* __restrict__ out, int E) {
  long t = (long)blockIdx.x * blockDim.x + threadIdx.x;
  if (t >= (long)E * 64) return;
  int e = (int)(t >> 6);
  int c = (int)(t & 63);
  int s = ei[e];
  int dd = ei[E + e];
  atomicAdd(&out[(size_t)dd * 64 + c], feat[(size_t)s * 64 + c] * w[e]);
}

// ---------------------------------------------------------------------------
// Layer 1: z = tanh([ [aggp/dp, x] @ w1p + b1p , [aggn/dn, x] @ w1n + b1n ])
// K = 128, 32 outputs per branch.  One wave per 16-node tile, 16 WMMAs.
// ---------------------------------------------------------------------------
__global__ __launch_bounds__(256) void layer1_kernel(
    const float* __restrict__ x,
    const float* __restrict__ aggp, const float* __restrict__ aggn,
    const float* __restrict__ dp,   const float* __restrict__ dn,
    const _Float16* __restrict__ pw1p, const float* __restrict__ b1p,
    const _Float16* __restrict__ pw1n, const float* __restrict__ b1n,
    float* __restrict__ z, int n) {
  int tile = (blockIdx.x * blockDim.x + threadIdx.x) >> 5;
  int lane = threadIdx.x & 31;
  if (tile * 16 >= n) return;
  int m   = lane & 15;
  int khi = lane >> 4;
  int rowA = tile * 16 + m;
  float invdp = 1.0f / fmaxf(dp[rowA], 1e-12f);
  float invdn = 1.0f / fmaxf(dn[rowA], 1e-12f);
  const float* ap_row = aggp + (size_t)rowA * 64;
  const float* an_row = aggn + (size_t)rowA * 64;
  const float* x_row  = x    + (size_t)rowA * 64;

  v8f accp0 = {}; v8f accp1 = {}; v8f accn0 = {}; v8f accn1 = {};
#pragma unroll
  for (int k0 = 0; k0 < 128; k0 += 32) {
    v16h fap, fan;
#pragma unroll
    for (int i = 0; i < 16; ++i) {
      int k = a_kmap(k0, i, khi);
      float vp, vn;
      if (k < 64) { vp = ap_row[k] * invdp; vn = an_row[k] * invdn; }
      else        { vp = x_row[k - 64];     vn = vp; }
      fap[i] = (_Float16)vp;
      fan[i] = (_Float16)vn;
    }
    int ks = k0 >> 5;
    v16h bp0 = load_packed_frag(pw1p, ks * 2 + 0, lane);
    v16h bp1 = load_packed_frag(pw1p, ks * 2 + 1, lane);
    v16h bn0 = load_packed_frag(pw1n, ks * 2 + 0, lane);
    v16h bn1 = load_packed_frag(pw1n, ks * 2 + 1, lane);
    accp0 = wmma_f32_f16(fap, bp0, accp0);
    accp1 = wmma_f32_f16(fap, bp1, accp1);
    accn0 = wmma_f32_f16(fan, bn0, accn0);
    accn1 = wmma_f32_f16(fan, bn1, accn1);
  }
  float bp0 = b1p[m], bp1 = b1p[m + 16], bn0 = b1n[m], bn1 = b1n[m + 16];
#pragma unroll
  for (int i = 0; i < 8; ++i) {
    int rowD = tile * 16 + i + khi * 8;
    float* zr = z + (size_t)rowD * 64;
    zr[m]      = tanhf(accp0[i] + bp0);
    zr[m + 16] = tanhf(accp1[i] + bp1);
    zr[m + 32] = tanhf(accn0[i] + bn0);
    zr[m + 48] = tanhf(accn1[i] + bn1);
  }
}

// ---------------------------------------------------------------------------
// Layer 2 (in-place on z):
//  op in: [aggp[:, 0:32]/dp, aggn[:,32:64]/dn, z[:,0:32]]  @ w2p + b2p
//  on in: [aggp[:,32:64]/dp, aggn[:, 0:32]/dn, z[:,32:64]] @ w2n + b2n
//  z <- tanh([op,on]).  K = 96, 12 WMMAs per tile.
// ---------------------------------------------------------------------------
__global__ __launch_bounds__(256) void layer2_kernel(
    float* z,
    const float* __restrict__ aggp, const float* __restrict__ aggn,
    const float* __restrict__ dp,   const float* __restrict__ dn,
    const _Float16* __restrict__ pw2p, const float* __restrict__ b2p,
    const _Float16* __restrict__ pw2n, const float* __restrict__ b2n,
    int n) {
  int tile = (blockIdx.x * blockDim.x + threadIdx.x) >> 5;
  int lane = threadIdx.x & 31;
  if (tile * 16 >= n) return;
  int m   = lane & 15;
  int khi = lane >> 4;
  int rowA = tile * 16 + m;
  float invdp = 1.0f / fmaxf(dp[rowA], 1e-12f);
  float invdn = 1.0f / fmaxf(dn[rowA], 1e-12f);
  const float* ap_row = aggp + (size_t)rowA * 64;
  const float* an_row = aggn + (size_t)rowA * 64;
  const float* z_row  = z    + (size_t)rowA * 64;

  v8f accp0 = {}; v8f accp1 = {}; v8f accn0 = {}; v8f accn1 = {};
#pragma unroll
  for (int k0 = 0; k0 < 96; k0 += 32) {
    v16h fap, fan;
#pragma unroll
    for (int i = 0; i < 16; ++i) {
      int k = a_kmap(k0, i, khi);
      float vp, vn;
      if (k < 32)      { vp = ap_row[k]      * invdp;   // wmean(zp,pos)
                         vn = ap_row[k + 32] * invdp; } // wmean(zn,pos)
      else if (k < 64) { vp = an_row[k]      * invdn;   // wmean(zn,neg) cols 32..63
                         vn = an_row[k - 32] * invdn; } // wmean(zp,neg) cols 0..31
      else             { vp = z_row[k - 64];            // zp
                         vn = z_row[k - 32]; }          // zn
      fap[i] = (_Float16)vp;
      fan[i] = (_Float16)vn;
    }
    int ks = k0 >> 5;
    v16h bp0 = load_packed_frag(pw2p, ks * 2 + 0, lane);
    v16h bp1 = load_packed_frag(pw2p, ks * 2 + 1, lane);
    v16h bn0 = load_packed_frag(pw2n, ks * 2 + 0, lane);
    v16h bn1 = load_packed_frag(pw2n, ks * 2 + 1, lane);
    accp0 = wmma_f32_f16(fap, bp0, accp0);
    accp1 = wmma_f32_f16(fap, bp1, accp1);
    accn0 = wmma_f32_f16(fan, bn0, accn0);
    accn1 = wmma_f32_f16(fan, bn1, accn1);
  }
  float bp0 = b2p[m], bp1 = b2p[m + 16], bn0 = b2n[m], bn1 = b2n[m + 16];
#pragma unroll
  for (int i = 0; i < 8; ++i) {
    int rowD = tile * 16 + i + khi * 8;
    float* zr = z + (size_t)rowD * 64;
    zr[m]      = tanhf(accp0[i] + bp0);
    zr[m + 16] = tanhf(accp1[i] + bp1);
    zr[m + 32] = tanhf(accn0[i] + bn0);
    zr[m + 48] = tanhf(accn1[i] + bn1);
  }
}

// ---------------------------------------------------------------------------
// Layer 3: out = tanh(z @ wout + bout).  K = 64, 64 cols, 8 WMMAs per tile.
// ---------------------------------------------------------------------------
__global__ __launch_bounds__(256) void layer3_kernel(
    const float* __restrict__ z,
    const _Float16* __restrict__ pwout, const float* __restrict__ bout,
    float* __restrict__ out, int n) {
  int tile = (blockIdx.x * blockDim.x + threadIdx.x) >> 5;
  int lane = threadIdx.x & 31;
  if (tile * 16 >= n) return;
  int m   = lane & 15;
  int khi = lane >> 4;
  int rowA = tile * 16 + m;
  const float* z_row = z + (size_t)rowA * 64;

  v8f acc[4] = {v8f{}, v8f{}, v8f{}, v8f{}};
#pragma unroll
  for (int k0 = 0; k0 < 64; k0 += 32) {
    v16h fa;
#pragma unroll
    for (int i = 0; i < 16; ++i) {
      fa[i] = (_Float16)z_row[a_kmap(k0, i, khi)];
    }
    int ks = k0 >> 5;
#pragma unroll
    for (int t = 0; t < 4; ++t) {
      v16h fb = load_packed_frag(pwout, ks * 4 + t, lane);
      acc[t] = wmma_f32_f16(fa, fb, acc[t]);
    }
  }
#pragma unroll
  for (int t = 0; t < 4; ++t) {
    float bb = bout[t * 16 + m];
#pragma unroll
    for (int i = 0; i < 8; ++i) {
      int rowD = tile * 16 + i + khi * 8;
      out[(size_t)rowD * 64 + t * 16 + m] = tanhf(acc[t][i] + bb);
    }
  }
}

// ---------------------------------------------------------------------------
extern "C" void kernel_launch(void* const* d_in, const int* in_sizes, int n_in,
                              void* d_out, int out_size, void* d_ws, size_t ws_size,
                              hipStream_t stream) {
  const int*   pos_ei = (const int*)  d_in[0];
  const int*   neg_ei = (const int*)  d_in[1];
  const float* pos_w  = (const float*)d_in[2];
  const float* neg_w  = (const float*)d_in[3];
  const float* x      = (const float*)d_in[4];
  const float* w1p    = (const float*)d_in[5];
  const float* b1p    = (const float*)d_in[6];
  const float* w1n    = (const float*)d_in[7];
  const float* b1n    = (const float*)d_in[8];
  const float* w2p    = (const float*)d_in[9];
  const float* b2p    = (const float*)d_in[10];
  const float* w2n    = (const float*)d_in[11];
  const float* b2n    = (const float*)d_in[12];
  const float* wout   = (const float*)d_in[13];
  const float* bout   = (const float*)d_in[14];

  const int N  = in_sizes[4] / 64;   // 100000
  const int EP = in_sizes[2];        // 1,000,000
  const int EN = in_sizes[3];        // 500,000

  // Workspace layout:
  //   [packed f16 weights: 18432 halfs, padded to 10240 floats]
  //   dp[N] | dn[N] | aggp[64N] | aggn[64N] | z[64N]
  _Float16* pw1p = (_Float16*)d_ws;        // 128*32 = 4096 halfs
  _Float16* pw1n = pw1p + 4096;            // 4096
  _Float16* pw2p = pw1n + 4096;            // 96*32 = 3072
  _Float16* pw2n = pw2p + 3072;            // 3072
  _Float16* pwout = pw2n + 3072;           // 64*64 = 4096
  float* fbase = (float*)d_ws + 10240;     // 40 KB in, 32B-aligned
  float* dp   = fbase;
  float* dn   = dp + N;
  float* aggp = dn + N;
  float* aggn = aggp + (size_t)64 * N;
  float* z    = aggn + (size_t)64 * N;
  float* out  = (float*)d_out;

  const dim3 blk(256);

  // ---- pack weights into WMMA B-fragment layout (tiny, once per launch) ---
  pack_w_kernel<<<(128 * 32 + 255) / 256, blk, 0, stream>>>(w1p, pw1p, 128, 32);
  pack_w_kernel<<<(128 * 32 + 255) / 256, blk, 0, stream>>>(w1n, pw1n, 128, 32);
  pack_w_kernel<<<( 96 * 32 + 255) / 256, blk, 0, stream>>>(w2p, pw2p,  96, 32);
  pack_w_kernel<<<( 96 * 32 + 255) / 256, blk, 0, stream>>>(w2n, pw2n,  96, 32);
  pack_w_kernel<<<( 64 * 64 + 255) / 256, blk, 0, stream>>>(wout, pwout, 64, 64);

  // ---- layer 1 aggregation ------------------------------------------------
  hipMemsetAsync(dp, 0, sizeof(float) * ((size_t)130 * N), stream); // dp,dn,aggp,aggn
  degree_kernel<<<(EP + 255) / 256, blk, 0, stream>>>(pos_ei, pos_w, dp, EP);
  degree_kernel<<<(EN + 255) / 256, blk, 0, stream>>>(neg_ei, neg_w, dn, EN);

  const long tp = (long)EP * 64, tn = (long)EN * 64;
  scatter_kernel<<<(unsigned)((tp + 255) / 256), blk, 0, stream>>>(pos_ei, pos_w, x, aggp, EP);
  scatter_kernel<<<(unsigned)((tn + 255) / 256), blk, 0, stream>>>(neg_ei, neg_w, x, aggn, EN);

  // ---- layer 1 dense (WMMA) ----------------------------------------------
  const int tiles    = (N + 15) / 16;
  const int l_blocks = (tiles + 7) / 8;  // 8 waves (tiles) per 256-thread block
  layer1_kernel<<<l_blocks, blk, 0, stream>>>(x, aggp, aggn, dp, dn,
                                              pw1p, b1p, pw1n, b1n, z, N);

  // ---- layer 2 aggregation (reuse agg buffers) ---------------------------
  hipMemsetAsync(aggp, 0, sizeof(float) * ((size_t)128 * N), stream);
  scatter_kernel<<<(unsigned)((tp + 255) / 256), blk, 0, stream>>>(pos_ei, pos_w, z, aggp, EP);
  scatter_kernel<<<(unsigned)((tn + 255) / 256), blk, 0, stream>>>(neg_ei, neg_w, z, aggn, EN);

  // ---- layer 2 dense (WMMA, in-place on z) -------------------------------
  layer2_kernel<<<l_blocks, blk, 0, stream>>>(z, aggp, aggn, dp, dn,
                                              pw2p, b2p, pw2n, b2n, N);

  // ---- output projection (WMMA) ------------------------------------------
  layer3_kernel<<<l_blocks, blk, 0, stream>>>(z, pwout, bout, out, N);
}
